// PeerRegularizationLayerAttBidir_64407329571370
// MI455X (gfx1250) — matrix operand order repaired and checked by reading, not compiled, not run on hardware
//
#include <hip/hip_runtime.h>

typedef unsigned short u16;
typedef __attribute__((ext_vector_type(16))) __bf16 v16bf;
typedef __attribute__((ext_vector_type(8)))  float  v8f;

#define NPIX 4096

__device__ __forceinline__ float bf2f(u16 h) {
    union { unsigned int u; float f; } c; c.u = ((unsigned int)h) << 16; return c.f;
}
__device__ __forceinline__ u16 f2bf(float f) {
    union { float f; unsigned int u; } c; c.f = f;
    unsigned int u = c.u;
    unsigned int lsb = (u >> 16) & 1u;
    u += 0x7fffu + lsb;            // round-to-nearest-even
    return (u16)(u >> 16);
}

union Frag { v16bf v; uint4 q[2]; };

// ---------------------------------------------------------------------------
// [C, 4096] f32 channel-major  ->  [4096, C] bf16 pixel-major (LDS transpose)
// grid: (4096/32, C/32), block: (32, 8)
// ---------------------------------------------------------------------------
__global__ __launch_bounds__(256)
void to_pix_bf16(const float* __restrict__ fmap, u16* __restrict__ pix, int C) {
    __shared__ float tile[32][33];
    const int n0 = blockIdx.x * 32, c0 = blockIdx.y * 32;
    const int tx = threadIdx.x, ty = threadIdx.y;
    #pragma unroll
    for (int i = 0; i < 32; i += 8)
        tile[ty + i][tx] = fmap[(size_t)(c0 + ty + i) * NPIX + n0 + tx];
    __syncthreads();
    #pragma unroll
    for (int i = 0; i < 32; i += 8)
        pix[(size_t)(n0 + ty + i) * C + c0 + tx] = f2bf(tile[tx][ty + i]);
}

// flat f32 -> bf16 (weights already [M,K] row-major)
__global__ void cvt_bf16(const float* __restrict__ in, u16* __restrict__ out, int n) {
    int i = blockIdx.x * 256 + threadIdx.x;
    if (i < n) out[i] = f2bf(in[i]);
}

// ---------------------------------------------------------------------------
// NT GEMM via WMMA:  S[i,j] = sum_k A[i,k] * Bw[j,k]      (A:[4096,K], Bw:[M,K])
// Each wave computes a 32x32 tile (2x2 WMMA sub-tiles): 4 WMMAs per 8 loads.
// block (32,4): 4 waves stacked along i -> block tile 128x32.
// grid (4096/128, M/32)
// mode 0: Sout[i*M+j] = acc                            (f32 score matrix)
// mode 1: Hout[i*M+j] = bf16(relu(BN(acc + bias[j])))  (fused MLP layer)
// ---------------------------------------------------------------------------
__global__ __launch_bounds__(128)
void gemm_nt_wmma(const u16* __restrict__ A, const u16* __restrict__ Bw,
                  int K, int M,
                  float* __restrict__ Sout, u16* __restrict__ Hout,
                  const float* __restrict__ bias, const float* __restrict__ gam,
                  const float* __restrict__ bet,  const float* __restrict__ mu,
                  const float* __restrict__ var,  int mode) {
    const int lane = threadIdx.x;
    const int row  = lane & 15;
    const int hi   = lane >> 4;
    const int i0   = (blockIdx.x * 4 + threadIdx.y) * 32;
    const int j0   = blockIdx.y * 32;

    const u16* arow0 = A  + (size_t)(i0 + row) * K;        // A frags: lane holds row (l&15)
    const u16* arow1 = A  + (size_t)(i0 + 16 + row) * K;
    const u16* brow0 = Bw + (size_t)(j0 + row) * K;        // B frags: lane holds col (l&15)
    const u16* brow1 = Bw + (size_t)(j0 + 16 + row) * K;

    v8f acc00 = {0.f,0.f,0.f,0.f,0.f,0.f,0.f,0.f};
    v8f acc01 = acc00, acc10 = acc00, acc11 = acc00;

    for (int kb = 0; kb < K; kb += 32) {
        Frag a0, a1, b0, b1;
        // A 16x32 bf16 layout: VGPR0-3 -> K = hi*8 + 0..7 ; VGPR4-7 -> K = 16 + hi*8 + 0..7
        a0.q[0] = *(const uint4*)(arow0 + kb + hi * 8);
        a0.q[1] = *(const uint4*)(arow0 + kb + 16 + hi * 8);
        a1.q[0] = *(const uint4*)(arow1 + kb + hi * 8);
        a1.q[1] = *(const uint4*)(arow1 + kb + 16 + hi * 8);
        // B 32x16 bf16 layout: VGPR0-3 -> K = hi*8 + 0..7 ; VGPR4-7 -> K = 16 + hi*8 + 0..7
        b0.q[0] = *(const uint4*)(brow0 + kb + hi * 8);
        b0.q[1] = *(const uint4*)(brow0 + kb + 16 + hi * 8);
        b1.q[0] = *(const uint4*)(brow1 + kb + hi * 8);
        b1.q[1] = *(const uint4*)(brow1 + kb + 16 + hi * 8);
        acc00 = __builtin_amdgcn_wmma_f32_16x16x32_bf16(false, a0.v, false, b0.v, (short)0, acc00, false, false);
        acc01 = __builtin_amdgcn_wmma_f32_16x16x32_bf16(false, a0.v, false, b1.v, (short)0, acc01, false, false);
        acc10 = __builtin_amdgcn_wmma_f32_16x16x32_bf16(false, a1.v, false, b0.v, (short)0, acc10, false, false);
        acc11 = __builtin_amdgcn_wmma_f32_16x16x32_bf16(false, a1.v, false, b1.v, (short)0, acc11, false, false);
    }

    // D layout: col = lane&15 (+16 per q), row = r + 8*hi (+16 per p)
    const int jq0 = j0 + row, jq1 = j0 + 16 + row;
    if (mode == 0) {
        #pragma unroll
        for (int r = 0; r < 8; ++r) {
            int ia = i0 + r + 8 * hi;
            int ib = ia + 16;
            Sout[(size_t)ia * M + jq0] = acc00[r];
            Sout[(size_t)ia * M + jq1] = acc01[r];
            Sout[(size_t)ib * M + jq0] = acc10[r];
            Sout[(size_t)ib * M + jq1] = acc11[r];
        }
    } else {
        const float sc0 = gam[jq0] * rsqrtf(var[jq0] + 1e-5f);
        const float sc1 = gam[jq1] * rsqrtf(var[jq1] + 1e-5f);
        const float o0 = bias[jq0] - mu[jq0], o1 = bias[jq1] - mu[jq1];
        const float bb0 = bet[jq0], bb1 = bet[jq1];
        #pragma unroll
        for (int r = 0; r < 8; ++r) {
            int ia = i0 + r + 8 * hi;
            int ib = ia + 16;
            Hout[(size_t)ia * M + jq0] = f2bf(fmaxf((acc00[r] + o0) * sc0 + bb0, 0.f));
            Hout[(size_t)ia * M + jq1] = f2bf(fmaxf((acc01[r] + o1) * sc1 + bb1, 0.f));
            Hout[(size_t)ib * M + jq0] = f2bf(fmaxf((acc10[r] + o0) * sc0 + bb0, 0.f));
            Hout[(size_t)ib * M + jq1] = f2bf(fmaxf((acc11[r] + o1) * sc1 + bb1, 0.f));
        }
    }
}

// row squared-norms of bf16 [4096,K]
__global__ __launch_bounds__(128)
void rownorm(const u16* __restrict__ X, int K, float* __restrict__ out) {
    __shared__ float red[128];
    const int i = blockIdx.x;
    float s = 0.f;
    for (int c = threadIdx.x; c < K; c += 128) {
        float v = bf2f(X[(size_t)i * K + c]); s += v * v;
    }
    red[threadIdx.x] = s; __syncthreads();
    for (int k = 64; k > 0; k >>= 1) {
        if (threadIdx.x < k) red[threadIdx.x] += red[threadIdx.x + k];
        __syncthreads();
    }
    if (threadIdx.x == 0) out[i] = red[0];
}

// "Conv1d(kernel=feat)" == dot + bias:  out[i] = sum_c X[i,c]*w[c] + b
__global__ void rowscore(const u16* __restrict__ X, int K, const float* __restrict__ w,
                         const float* __restrict__ b, float* __restrict__ out) {
    int i = blockIdx.x * 256 + threadIdx.x;
    if (i >= NPIX) return;
    float s = b[0];
    for (int c = 0; c < K; ++c) s += bf2f(X[(size_t)i * K + c]) * w[c];
    out[i] = s;
}

// ---------------------------------------------------------------------------
// Hard-kNN softmax attention, one 256-thread block per query row.
//   d2[j] = max(0, nq[i]+nk[j]-2*S[i,j]); kth = 5th order statistic;
//   keep d2<=kth; w = exp(softplus(q_i+k_j)-max); out = (w·vals)/sum(w)
// vals channel-major [V,4096]; out_fmap channel-major region of d_out.
// ---------------------------------------------------------------------------
#define MAXL 64
__global__ __launch_bounds__(256)
void attend(const float* __restrict__ S, const float* __restrict__ nq,
            const float* __restrict__ nk, const float* __restrict__ qs,
            const float* __restrict__ ks, const float* __restrict__ vals, int V,
            float* __restrict__ out_fmap, u16* __restrict__ out_pixb) {
    const int i = blockIdx.x, tid = threadIdx.x;
    __shared__ float rv[256]; __shared__ int ri[256];
    __shared__ int sel[5];    __shared__ float sh_kth;
    __shared__ int cnt;       __shared__ int lj[MAXL]; __shared__ float lw[MAXL];
    __shared__ float sh_amax, sh_sumw;

    const float myn = nq[i];
    float d2loc[16];
    #pragma unroll
    for (int t = 0; t < 16; ++t) {
        int j = tid + 256 * t;
        d2loc[t] = fmaxf(myn + nk[j] - 2.f * S[(size_t)i * NPIX + j], 0.f);
    }
    // 5 rounds of argmin with exclusion -> exact 5th smallest (tie-safe)
    for (int r = 0; r < 5; ++r) {
        float bv = 3.4e38f; int bj = 1 << 30;
        #pragma unroll
        for (int t = 0; t < 16; ++t) {
            int j = tid + 256 * t;
            bool skip = false;
            for (int s = 0; s < r; ++s) skip |= (j == sel[s]);
            float v = d2loc[t];
            if (!skip && (v < bv || (v == bv && j < bj))) { bv = v; bj = j; }
        }
        rv[tid] = bv; ri[tid] = bj; __syncthreads();
        for (int k = 128; k > 0; k >>= 1) {
            if (tid < k && (rv[tid + k] < rv[tid] ||
                (rv[tid + k] == rv[tid] && ri[tid + k] < ri[tid]))) {
                rv[tid] = rv[tid + k]; ri[tid] = ri[tid + k];
            }
            __syncthreads();
        }
        if (tid == 0) { sel[r] = ri[0]; if (r == 4) sh_kth = rv[0]; }
        __syncthreads();
    }
    const float kth = sh_kth;
    if (tid == 0) cnt = 0;
    __syncthreads();

    const float qi = qs[i];
    float amax = -3.4e38f;
    #pragma unroll
    for (int t = 0; t < 16; ++t) {
        int j = tid + 256 * t;
        if (d2loc[t] <= kth) {
            float x = qi + ks[j];
            float a = (x > 20.f) ? x : log1pf(expf(x));   // softplus
            amax = fmaxf(amax, a);
            int p = atomicAdd(&cnt, 1);
            if (p < MAXL) lj[p] = j;
        }
    }
    rv[tid] = amax; __syncthreads();
    for (int k = 128; k > 0; k >>= 1) {
        if (tid < k) rv[tid] = fmaxf(rv[tid], rv[tid + k]);
        __syncthreads();
    }
    if (tid == 0) sh_amax = rv[0];
    __syncthreads();

    const int nl = min(cnt, MAXL);
    for (int l = tid; l < nl; l += 256) {
        float x = qi + ks[lj[l]];
        float a = (x > 20.f) ? x : log1pf(expf(x));
        lw[l] = expf(a - sh_amax);
    }
    __syncthreads();
    if (tid == 0) { float s = 0.f; for (int l = 0; l < nl; ++l) s += lw[l]; sh_sumw = s; }
    __syncthreads();

    const float inv = 1.f / sh_sumw;
    for (int c = tid; c < V; c += 256) {
        float acc = 0.f;
        for (int l = 0; l < nl; ++l) acc += lw[l] * vals[(size_t)c * NPIX + lj[l]];
        float o = acc * inv;
        out_fmap[(size_t)c * NPIX + i] = o;
        if (out_pixb) out_pixb[(size_t)i * V + c] = f2bf(o);
    }
}

// ---------------------------------------------------------------------------
extern "C" void kernel_launch(void* const* d_in, const int* in_sizes, int n_in,
                              void* d_out, int out_size, void* d_ws, size_t ws_size,
                              hipStream_t stream) {
    (void)in_sizes; (void)n_in; (void)out_size; (void)ws_size;
    const float* inp_cont    = (const float*)d_in[0];
    const float* peers_cont  = (const float*)d_in[2];
    const float* peers_style = (const float*)d_in[3];
    const float *w1c = (const float*)d_in[4],  *b1c = (const float*)d_in[5],
                *g1c = (const float*)d_in[6],  *be1c = (const float*)d_in[7],
                *m1c = (const float*)d_in[8],  *v1c = (const float*)d_in[9];
    const float *w2c = (const float*)d_in[10], *b2c = (const float*)d_in[11],
                *g2c = (const float*)d_in[12], *be2c = (const float*)d_in[13],
                *m2c = (const float*)d_in[14], *v2c = (const float*)d_in[15];
    const float *c1cw = (const float*)d_in[16], *c1cb = (const float*)d_in[17],
                *c2cw = (const float*)d_in[18], *c2cb = (const float*)d_in[19];
    const float *w1s = (const float*)d_in[20], *b1s = (const float*)d_in[21],
                *g1s = (const float*)d_in[22], *be1s = (const float*)d_in[23],
                *m1s = (const float*)d_in[24], *v1s = (const float*)d_in[25];
    const float *w2s = (const float*)d_in[26], *b2s = (const float*)d_in[27],
                *g2s = (const float*)d_in[28], *be2s = (const float*)d_in[29],
                *m2s = (const float*)d_in[30], *v2s = (const float*)d_in[31];
    const float *c1sw = (const float*)d_in[32], *c1sb = (const float*)d_in[33],
                *c2sw = (const float*)d_in[34], *c2sb = (const float*)d_in[35];
    float* out = (float*)d_out;

    char* ws = (char*)d_ws; size_t off = 0;
    auto alloc = [&](size_t bytes) -> void* {
        off = (off + 255) & ~(size_t)255;
        void* p = ws + off; off += bytes; return p;
    };
    u16* ipb     = (u16*)alloc((size_t)NPIX * 256 * 2);
    u16* ppb     = (u16*)alloc((size_t)NPIX * 256 * 2);
    u16* psb     = (u16*)alloc((size_t)NPIX * 512 * 2);
    u16* outstyb = (u16*)alloc((size_t)NPIX * 512 * 2);
    u16* w1cb    = (u16*)alloc(128 * 256 * 2);
    u16* w2cb    = (u16*)alloc(64 * 128 * 2);
    u16* w1sb    = (u16*)alloc(256 * 512 * 2);
    u16* w2sb    = (u16*)alloc(128 * 256 * 2);
    u16* t1b     = (u16*)alloc((size_t)NPIX * 256 * 2);
    u16* aqb     = (u16*)alloc((size_t)NPIX * 64 * 2);
    u16* akb     = (u16*)alloc((size_t)NPIX * 64 * 2);
    u16* sqb     = (u16*)alloc((size_t)NPIX * 128 * 2);
    u16* skb     = (u16*)alloc((size_t)NPIX * 128 * 2);
    float* qsv   = (float*)alloc(NPIX * 4);
    float* ksv   = (float*)alloc(NPIX * 4);
    float* nqv   = (float*)alloc(NPIX * 4);
    float* nkv   = (float*)alloc(NPIX * 4);
    float* Sbuf  = (float*)alloc((size_t)NPIX * NPIX * 4);

    const dim3 tb(32, 8);
    to_pix_bf16<<<dim3(128, 8),  tb, 0, stream>>>(inp_cont,    ipb, 256);
    to_pix_bf16<<<dim3(128, 8),  tb, 0, stream>>>(peers_cont,  ppb, 256);
    to_pix_bf16<<<dim3(128, 16), tb, 0, stream>>>(peers_style, psb, 512);
    cvt_bf16<<<(128 * 256 + 255) / 256, 256, 0, stream>>>(w1c, w1cb, 128 * 256);
    cvt_bf16<<<(64  * 128 + 255) / 256, 256, 0, stream>>>(w2c, w2cb, 64 * 128);
    cvt_bf16<<<(256 * 512 + 255) / 256, 256, 0, stream>>>(w1s, w1sb, 256 * 512);
    cvt_bf16<<<(128 * 256 + 255) / 256, 256, 0, stream>>>(w2s, w2sb, 128 * 256);

    const dim3 gb(32, 4);   // 4 waves, block tile 128(i) x 32(j)
    // ---- stage 1: content-space kNN gathers peer style ----
    gemm_nt_wmma<<<dim3(32, 4),   gb, 0, stream>>>(ipb, w1cb, 256, 128, nullptr, t1b, b1c, g1c, be1c, m1c, v1c, 1);
    gemm_nt_wmma<<<dim3(32, 2),   gb, 0, stream>>>(t1b, w2cb, 128, 64,  nullptr, aqb, b2c, g2c, be2c, m2c, v2c, 1);
    gemm_nt_wmma<<<dim3(32, 4),   gb, 0, stream>>>(ppb, w1cb, 256, 128, nullptr, t1b, b1c, g1c, be1c, m1c, v1c, 1);
    gemm_nt_wmma<<<dim3(32, 2),   gb, 0, stream>>>(t1b, w2cb, 128, 64,  nullptr, akb, b2c, g2c, be2c, m2c, v2c, 1);
    rowscore<<<16, 256, 0, stream>>>(aqb, 64, c1cw, c1cb, qsv);
    rowscore<<<16, 256, 0, stream>>>(akb, 64, c2cw, c2cb, ksv);
    rownorm<<<NPIX, 128, 0, stream>>>(ipb, 256, nqv);
    rownorm<<<NPIX, 128, 0, stream>>>(ppb, 256, nkv);
    gemm_nt_wmma<<<dim3(32, 128), gb, 0, stream>>>(ipb, ppb, 256, NPIX, Sbuf, nullptr,
                                                   nullptr, nullptr, nullptr, nullptr, nullptr, 0);
    attend<<<NPIX, 256, 0, stream>>>(Sbuf, nqv, nkv, qsv, ksv, peers_style, 512,
                                     out + (size_t)256 * NPIX, outstyb);

    // ---- stage 2: style-space kNN gathers peer content ----
    gemm_nt_wmma<<<dim3(32, 8),   gb, 0, stream>>>(outstyb, w1sb, 512, 256, nullptr, t1b, b1s, g1s, be1s, m1s, v1s, 1);
    gemm_nt_wmma<<<dim3(32, 4),   gb, 0, stream>>>(t1b,     w2sb, 256, 128, nullptr, sqb, b2s, g2s, be2s, m2s, v2s, 1);
    gemm_nt_wmma<<<dim3(32, 8),   gb, 0, stream>>>(psb,     w1sb, 512, 256, nullptr, t1b, b1s, g1s, be1s, m1s, v1s, 1);
    gemm_nt_wmma<<<dim3(32, 4),   gb, 0, stream>>>(t1b,     w2sb, 256, 128, nullptr, skb, b2s, g2s, be2s, m2s, v2s, 1);
    rowscore<<<16, 256, 0, stream>>>(sqb, 128, c1sw, c1sb, qsv);
    rowscore<<<16, 256, 0, stream>>>(skb, 128, c2sw, c2sb, ksv);
    rownorm<<<NPIX, 128, 0, stream>>>(outstyb, 512, nqv);
    rownorm<<<NPIX, 128, 0, stream>>>(psb,     512, nkv);
    gemm_nt_wmma<<<dim3(32, 128), gb, 0, stream>>>(outstyb, psb, 512, NPIX, Sbuf, nullptr,
                                                   nullptr, nullptr, nullptr, nullptr, nullptr, 0);
    attend<<<NPIX, 256, 0, stream>>>(Sbuf, nqv, nkv, qsv, ksv, peers_cont, 256,
                                     out, nullptr);
}
